// TritonBlockSparseAttention_31817117729142
// MI455X (gfx1250) — matrix operand order repaired and checked by reading, not compiled, not run on hardware
//
#include <hip/hip_runtime.h>

typedef __bf16 bf16;
typedef __attribute__((ext_vector_type(16))) __bf16 v16bf;
typedef __attribute__((ext_vector_type(8)))  __bf16 v8bf;
typedef __attribute__((ext_vector_type(8)))  float   v8f;
typedef __attribute__((ext_vector_type(4)))  float   v4f;

#define BATCH 8
#define SEQ   4096
#define DIM   64
#define BLK   128
#define NB    32

// One workgroup handles one (batch, query-block-row). 256 threads = 8 wave32;
// wave w owns queries [qb*128 + w*16, +16).
__global__ __launch_bounds__(256, 1)
void bsattn_kernel(const float* __restrict__ q, const float* __restrict__ kmat,
                   const float* __restrict__ vmat, const int* __restrict__ layout,
                   float* __restrict__ out) {
  __shared__ __align__(32) bf16 ldsK[BLK * DIM];   // [k_row][d]   row-major
  __shared__ __align__(32) bf16 ldsVT[DIM * BLK];  // [d][k_row]   transposed

  const int b    = blockIdx.x / NB;
  const int qb   = blockIdx.x % NB;
  const int tid  = threadIdx.x;
  const int lane = tid & 31;
  const int wave = tid >> 5;
  const int lmod  = lane & 15;
  const int lhalf = lane >> 4;

  // ---- resident Q^T B-fragments (pre-scaled by 1/sqrt(D) = 0.125) ----
  // B-frag (32x16): lane -> column q = lmod; element e -> K(d) = lhalf*16 + e
  const float* qrow = q + ((size_t)b * SEQ + (size_t)qb * BLK + wave * 16 + lmod) * DIM;
  v16bf qf[2];
  #pragma unroll
  for (int c = 0; c < 2; ++c) {
    const int d0 = c * 32 + lhalf * 16;
    #pragma unroll
    for (int e = 0; e < 16; ++e)
      qf[c][e] = (bf16)(qrow[d0 + e] * 0.125f);
  }

  v8f acc_o[4];                 // 16q x 64d output accumulator (4 d-tiles)
  #pragma unroll
  for (int t = 0; t < 4; ++t) { v8f z{}; acc_o[t] = z; }
  float m_i = -INFINITY;        // running max for query q = lmod
  float l_i = 0.0f;             // running denom for query q = lmod

  for (int kb = 0; kb < NB; ++kb) {
    if (layout[qb * NB + kb] == 0) continue;   // uniform scalar branch

    __syncthreads();   // previous iteration's LDS readers done
    {   // cooperative f32->bf16 stage of K block (row-major) and V block (transposed)
      const float* kg = kmat + ((size_t)b * SEQ + (size_t)kb * BLK) * DIM;
      const float* vg = vmat + ((size_t)b * SEQ + (size_t)kb * BLK) * DIM;
      for (int i = tid; i < BLK * (DIM / 4); i += 256) {
        const int row = i >> 4;
        const int d4  = (i & 15) << 2;
        const v4f kv = *(const v4f*)(kg + row * DIM + d4);
        const v4f vv = *(const v4f*)(vg + row * DIM + d4);
        #pragma unroll
        for (int j = 0; j < 4; ++j) {
          ldsK[row * DIM + d4 + j]      = (bf16)kv[j];
          ldsVT[(d4 + j) * BLK + row]   = (bf16)vv[j];
        }
      }
    }
    __syncthreads();

    // ---- S^T = (K * Q^T): 8 tiles of 16k x 16q per wave ----
    // lane holds q = lmod; k values = { t*16 + r + 8*lhalf : r in 0..7 }
    v8f sc[8];
    #pragma unroll
    for (int t = 0; t < 8; ++t) {
      v8f c{};
      #pragma unroll
      for (int ch = 0; ch < 2; ++ch) {
        // A-frag (16x32): lane -> row k = t*16+lmod; elems: d = ch*32 + lhalf*16 + {0..15}
        const int krow = t * 16 + lmod;
        const int d0   = ch * 32 + lhalf * 16;
        const v8bf lo = *(const v8bf*)(&ldsK[krow * DIM + d0]);
        const v8bf hi = *(const v8bf*)(&ldsK[krow * DIM + d0 + 8]);
        v16bf a;
        #pragma unroll
        for (int e = 0; e < 8; ++e) { a[e] = lo[e]; a[e + 8] = hi[e]; }
        c = __builtin_amdgcn_wmma_f32_16x16x32_bf16(false, a, false, qf[ch],
                                                    (short)0, c, false, false);
      }
      sc[t] = c;
    }

    // ---- online softmax: per lane 64 scores of query lmod; pair lane^16 has the rest ----
    float mt = -INFINITY;
    #pragma unroll
    for (int t = 0; t < 8; ++t)
      #pragma unroll
      for (int r = 0; r < 8; ++r) mt = fmaxf(mt, sc[t][r]);
    mt = fmaxf(mt, __shfl_xor(mt, 16, 32));
    const float m_new = fmaxf(m_i, mt);
    const float alpha = __expf(m_i - m_new);
    float lsum = 0.0f;
    #pragma unroll
    for (int t = 0; t < 8; ++t)
      #pragma unroll
      for (int r = 0; r < 8; ++r) {
        const float p = __expf(sc[t][r] - m_new);
        sc[t][r] = p;
        lsum += p;
      }
    lsum += __shfl_xor(lsum, 16, 32);
    l_i = l_i * alpha + lsum;
    m_i = m_new;

    // rescale acc_o: its row q = r + 8*lhalf, so fetch alpha from lane (r + 8*lhalf)
    #pragma unroll
    for (int r = 0; r < 8; ++r) {
      const float ar = __shfl(alpha, r + lhalf * 8, 32);
      #pragma unroll
      for (int t = 0; t < 4; ++t) acc_o[t][r] *= ar;
    }

    // ---- O += P(16q x 128k) * V(128k x 64d) ----
    // Sc tiles are already in PV A-frag arrangement: e<8 -> tile 2kc vgpr e; e>=8 -> tile 2kc+1
    #pragma unroll
    for (int kc = 0; kc < 4; ++kc) {
      v16bf a;
      #pragma unroll
      for (int e = 0; e < 8; ++e) {
        a[e]     = (bf16)sc[2 * kc][e];
        a[e + 8] = (bf16)sc[2 * kc + 1][e];
      }
      #pragma unroll
      for (int t = 0; t < 4; ++t) {
        // B-frag (32x16): lane -> col d = t*16+lmod; elems: k = kc*32 + lhalf*8 + {0..7, 16..23}
        const v8bf blo = *(const v8bf*)(&ldsVT[(t * 16 + lmod) * BLK + kc * 32 + lhalf * 8]);
        const v8bf bhi = *(const v8bf*)(&ldsVT[(t * 16 + lmod) * BLK + kc * 32 + lhalf * 8 + 16]);
        v16bf bvt;
        #pragma unroll
        for (int e = 0; e < 8; ++e) { bvt[e] = blo[e]; bvt[e + 8] = bhi[e]; }
        acc_o[t] = __builtin_amdgcn_wmma_f32_16x16x32_bf16(false, a, false, bvt,
                                                           (short)0, acc_o[t], false, false);
      }
    }
  }

  // ---- epilogue: divide by softmax denom and store f32 ----
  const float inv = 1.0f / l_i;     // keyed to q = lmod
  float* obase = out + ((size_t)b * SEQ + (size_t)qb * BLK + wave * 16) * DIM;
  #pragma unroll
  for (int r = 0; r < 8; ++r) {
    const float ir  = __shfl(inv, r + lhalf * 8, 32);   // denom for row q = r + 8*lhalf
    const int  qr   = r + lhalf * 8;
    #pragma unroll
    for (int t = 0; t < 4; ++t)
      obase[qr * DIM + t * 16 + lmod] = acc_o[t][r] * ir;
  }
}

extern "C" void kernel_launch(void* const* d_in, const int* in_sizes, int n_in,
                              void* d_out, int out_size, void* d_ws, size_t ws_size,
                              hipStream_t stream) {
  const float* q      = (const float*)d_in[0];
  const float* k      = (const float*)d_in[1];
  const float* v      = (const float*)d_in[2];
  const int*   layout = (const int*)d_in[3];
  float*       out    = (float*)d_out;

  dim3 grid(BATCH * NB);   // 256 workgroups: one per (batch, query block-row)
  dim3 block(256);         // 8 wave32
  bsattn_kernel<<<grid, block, 0, stream>>>(q, k, v, layout, out);
}
